// GPT_292057776335
// MI455X (gfx1250) — matrix operand order repaired and checked by reading, not compiled, not run on hardware
//
#include <hip/hip_runtime.h>
#include <hip/hip_bf16.h>
#include <cstdint>

#define VOCAB  50257
#define VPAD   50304   // VOCAB padded to multiple of 128 for lm_w^T
#define DMODEL 768
#define NHEAD  12
#define NLAYER 2
#define BATCH  4
#define SEQ    1024
#define HDIM   64
#define EPSLN  1e-5f

typedef __attribute__((ext_vector_type(16))) _Float16 v16h;
typedef __attribute__((ext_vector_type(8)))  _Float16 v8h;
typedef __attribute__((ext_vector_type(8)))  float    v8f;

static __device__ __forceinline__ v16h hcat(v8h a, v8h b) {
  return __builtin_shufflevector(a, b, 0,1,2,3,4,5,6,7,8,9,10,11,12,13,14,15);
}
static __device__ __forceinline__ v8h ld8h(const _Float16* p) { return *(const v8h*)p; }

static __device__ __forceinline__ v8f wmma_f16(v16h a, v16h b, v8f c) {
  // (neg_a, A, neg_b, B, c_mod, C, reuse_a, reuse_b)
  return __builtin_amdgcn_wmma_f32_16x16x32_f16(false, a, false, b, (short)0, c, false, false);
}
static __device__ __forceinline__ v8f zero8f() {
  v8f z;
#pragma unroll
  for (int i = 0; i < 8; i++) z[i] = 0.f;
  return z;
}

// ---------------- embedding: x[b,t,:] = wte[idx[b,t],:] + wpe[t,:] (f32) ----
__global__ __launch_bounds__(256) void embed_kernel(const int* __restrict__ idx,
    const float* __restrict__ wte, const float* __restrict__ wpe,
    float* __restrict__ x) {
  const int bt  = blockIdx.x;
  const int t   = bt % SEQ;
  const int tok = idx[bt];
  const int tid = threadIdx.x;
#pragma unroll
  for (int i = 0; i < 3; i++) {
    const int d = tid + 256 * i;
    x[(size_t)bt * DMODEL + d] =
        wte[(size_t)tok * DMODEL + d] + wpe[(size_t)t * DMODEL + d];
  }
}

// ---------------- layernorm over D=768 (f32 in, f16 out) --------------------
__global__ __launch_bounds__(256) void ln_kernel(const float* __restrict__ x,
    const float* __restrict__ gg, const float* __restrict__ bb,
    _Float16* __restrict__ out) {
  __shared__ float red[256];
  const int row = blockIdx.x;
  const int tid = threadIdx.x;
  const float* xr = x + (size_t)row * DMODEL;
  float v[3];
#pragma unroll
  for (int i = 0; i < 3; i++) v[i] = xr[tid + 256 * i];

  float s = v[0] + v[1] + v[2];
  red[tid] = s; __syncthreads();
  for (int off = 128; off > 0; off >>= 1) {
    if (tid < off) red[tid] += red[tid + off];
    __syncthreads();
  }
  const float mu = red[0] * (1.f / DMODEL);
  __syncthreads();

  float sq = 0.f;
#pragma unroll
  for (int i = 0; i < 3; i++) { const float d = v[i] - mu; sq += d * d; }
  red[tid] = sq; __syncthreads();
  for (int off = 128; off > 0; off >>= 1) {
    if (tid < off) red[tid] += red[tid + off];
    __syncthreads();
  }
  const float var = red[0] * (1.f / DMODEL);
  const float rstd = rsqrtf(var + EPSLN);
#pragma unroll
  for (int i = 0; i < 3; i++) {
    const int c = tid + 256 * i;
    out[(size_t)row * DMODEL + c] = (_Float16)((v[i] - mu) * rstd * gg[c] + bb[c]);
  }
}

// ------------ weight convert+transpose: f32 [K][N] -> f16 [Npad][K] ---------
__global__ __launch_bounds__(256) void convT_kernel(const float* __restrict__ in,
    _Float16* __restrict__ out, int K, int N) {
  __shared__ _Float16 s[32][33];
  const int tx = threadIdx.x & 31;
  const int ty = threadIdx.x >> 5;
  const int n0 = blockIdx.x * 32;
  const int k0 = blockIdx.y * 32;
#pragma unroll
  for (int i = 0; i < 4; i++) {
    const int k = k0 + ty + 8 * i;
    const int n = n0 + tx;
    const float v = (n < N) ? in[(size_t)k * N + n] : 0.f;
    s[ty + 8 * i][tx] = (_Float16)v;
  }
  __syncthreads();
#pragma unroll
  for (int i = 0; i < 4; i++) {
    const int n = n0 + ty + 8 * i;
    out[(size_t)n * K + k0 + tx] = s[tx][ty + 8 * i];
  }
}

// ---------------- LDS-free WMMA GEMM ----------------------------------------
// out[M,N] = A[M,K](f16) @ Wt[N,K]^T(f16)  (+bias, +gelu, +residual)
// 256 threads = 8 waves, block tile 128x128, wave tile 32x64, K-step 32.
// All operands read as WMMA fragments via 16B global loads (L2/WGP$-resident).
template <int ACT, int RES, int HASB, int OUTH>
__global__ __launch_bounds__(256) void gemm16_kernel(
    const _Float16* __restrict__ A, const _Float16* __restrict__ Wt,
    const float* __restrict__ bias, const float* __restrict__ res,
    float* __restrict__ outf, _Float16* __restrict__ outh, int N, int K) {
  const int tid  = threadIdx.x;
  const int lane = tid & 31;
  const int wv   = tid >> 5;
  const int g    = lane >> 4;
  const int ln   = lane & 15;
  const int m0   = blockIdx.y * 128;
  const int n0   = blockIdx.x * 128;
  const int wm   = wv & 3;
  const int wn   = wv >> 2;

  v8f acc[2][4];
#pragma unroll
  for (int mf = 0; mf < 2; mf++)
#pragma unroll
    for (int nf = 0; nf < 4; nf++) acc[mf][nf] = zero8f();

  const _Float16* arow = A  + (size_t)(m0 + wm * 32 + ln) * K;
  const _Float16* brow = Wt + (size_t)(n0 + wn * 64 + ln) * K;

  for (int k0 = 0; k0 < K; k0 += 32) {
    __builtin_prefetch(arow + k0 + 256, 0, 1);
    __builtin_prefetch(brow + k0 + 256, 0, 1);
    v16h af[2], bf[4];
#pragma unroll
    for (int mf = 0; mf < 2; mf++) {
      const _Float16* p = arow + (size_t)mf * 16 * K + k0;
      af[mf] = hcat(ld8h(p + 16 * g), ld8h(p + 16 * g + 8));
    }
#pragma unroll
    for (int nf = 0; nf < 4; nf++) {
      const _Float16* p = brow + (size_t)nf * 16 * K + k0 + 16 * g;
      bf[nf] = hcat(ld8h(p), ld8h(p + 8));
    }
#pragma unroll
    for (int mf = 0; mf < 2; mf++)
#pragma unroll
      for (int nf = 0; nf < 4; nf++)
        acc[mf][nf] = wmma_f16(af[mf], bf[nf], acc[mf][nf]);
  }

  // C layout: row = r + 8*g, col = lane%16
#pragma unroll
  for (int mf = 0; mf < 2; mf++) {
#pragma unroll
    for (int nf = 0; nf < 4; nf++) {
      const int col = n0 + wn * 64 + nf * 16 + ln;
      if (col < N) {
        const float bv = HASB ? bias[col] : 0.f;
#pragma unroll
        for (int r = 0; r < 8; r++) {
          const int row = m0 + wm * 32 + mf * 16 + r + 8 * g;
          float v = acc[mf][nf][r] + bv;
          if (ACT == 1) {
            const float u = v;
            v = 0.5f * u * (1.f + tanhf(0.7978845608f * (u + 0.044715f * u * u * u)));
          }
          if (RES) v += res[(size_t)row * N + col];
          if (OUTH) outh[(size_t)row * N + col] = (_Float16)v;
          else      outf[(size_t)row * N + col] = v;
        }
      }
    }
  }
}

// ---------------- QKV GEMM with attention-layout f16 epilogue ---------------
// qh/kh: [b][h][t][64] f16 ; vth: [b][h][64][t] f16 (pre-transposed for P@V)
__global__ __launch_bounds__(256) void qkv_gemm_kernel(
    const _Float16* __restrict__ A, const _Float16* __restrict__ Wt,
    const float* __restrict__ bias,
    _Float16* __restrict__ qh, _Float16* __restrict__ kh,
    _Float16* __restrict__ vth) {
  const int K = DMODEL, N = 3 * DMODEL;
  const int tid  = threadIdx.x;
  const int lane = tid & 31;
  const int wv   = tid >> 5;
  const int g    = lane >> 4;
  const int ln   = lane & 15;
  const int m0   = blockIdx.y * 128;
  const int n0   = blockIdx.x * 128;
  const int wm   = wv & 3;
  const int wn   = wv >> 2;

  v8f acc[2][4];
#pragma unroll
  for (int mf = 0; mf < 2; mf++)
#pragma unroll
    for (int nf = 0; nf < 4; nf++) acc[mf][nf] = zero8f();

  const _Float16* arow = A  + (size_t)(m0 + wm * 32 + ln) * K;
  const _Float16* brow = Wt + (size_t)(n0 + wn * 64 + ln) * K;

  for (int k0 = 0; k0 < K; k0 += 32) {
    v16h af[2], bf[4];
#pragma unroll
    for (int mf = 0; mf < 2; mf++) {
      const _Float16* p = arow + (size_t)mf * 16 * K + k0;
      af[mf] = hcat(ld8h(p + 16 * g), ld8h(p + 16 * g + 8));
    }
#pragma unroll
    for (int nf = 0; nf < 4; nf++) {
      const _Float16* p = brow + (size_t)nf * 16 * K + k0 + 16 * g;
      bf[nf] = hcat(ld8h(p), ld8h(p + 8));
    }
#pragma unroll
    for (int mf = 0; mf < 2; mf++)
#pragma unroll
      for (int nf = 0; nf < 4; nf++)
        acc[mf][nf] = wmma_f16(af[mf], bf[nf], acc[mf][nf]);
  }

  const int b  = m0 >> 10;           // 128 | 1024 -> whole block in one batch
  const int tb = m0 & 1023;
#pragma unroll
  for (int mf = 0; mf < 2; mf++) {
#pragma unroll
    for (int nf = 0; nf < 4; nf++) {
      const int col   = n0 + wn * 64 + nf * 16 + ln;
      const float bv  = bias[col];
      const int which = col / DMODEL;       // 0=q 1=k 2=v (16|64 -> lane-uniform)
      const int cc    = col - which * DMODEL;
      const int hh    = cc >> 6;
      const int dd    = cc & 63;
      const size_t hb = (size_t)b * NHEAD + hh;
      if (which == 2) {
        v8h pk;
#pragma unroll
        for (int r = 0; r < 8; r++) pk[r] = (_Float16)(acc[mf][nf][r] + bv);
        const int t0 = tb + wm * 32 + mf * 16 + 8 * g;   // 8 consecutive t
        *(v8h*)(vth + (hb * HDIM + dd) * SEQ + t0) = pk;
      } else {
        _Float16* dst = (which == 0 ? qh : kh) + hb * SEQ * HDIM + dd;
#pragma unroll
        for (int r = 0; r < 8; r++) {
          const int t = tb + wm * 32 + mf * 16 + r + 8 * g;
          dst[(size_t)t * HDIM] = (_Float16)(acc[mf][nf][r] + bv);
        }
      }
    }
  }
}

// ---------------- flash attention (causal, WMMA, zero-staging operands) -----
// grid (SEQ/64, B*H), 128 threads = 4 waves, each wave owns 16 q rows.
__global__ __launch_bounds__(128) void attn_kernel(
    const _Float16* __restrict__ qh, const _Float16* __restrict__ kh,
    const _Float16* __restrict__ vth, _Float16* __restrict__ y) {
  __shared__ __align__(16) _Float16 smem[4 * 640];  // per wave: P[16][40]
  const int tid  = threadIdx.x;
  const int lane = tid & 31;
  const int wv   = tid >> 5;
  const int g    = lane >> 4;
  const int ln   = lane & 15;
  const int bh   = blockIdx.y;
  const int b    = bh / NHEAD;
  const int h    = bh % NHEAD;
  const int q0   = blockIdx.x * 64 + wv * 16;

  const size_t hb = (size_t)b * NHEAD + h;
  const _Float16* qp = qh  + hb * SEQ * HDIM;
  const _Float16* kp = kh  + hb * SEQ * HDIM;
  const _Float16* vp = vth + hb * HDIM * SEQ;
  _Float16* ps = smem + wv * 640;

  // Q fragments (A layout) straight from global
  v16h qa[2];
  {
    const _Float16* qr = qp + (size_t)(q0 + ln) * HDIM;
#pragma unroll
    for (int j = 0; j < 2; j++)
      qa[j] = hcat(ld8h(qr + j * 32 + 16 * g), ld8h(qr + j * 32 + 8 + 16 * g));
  }

  v8f o[4];
#pragma unroll
  for (int nf = 0; nf < 4; nf++) o[nf] = zero8f();
  float mrow[8], lrow[8];
#pragma unroll
  for (int r = 0; r < 8; r++) { mrow[r] = -1e30f; lrow[r] = 0.f; }

  const int ktiles = (q0 + 15) / 32 + 1;  // wave-uniform -> EXEC all-1s for WMMA
  for (int kt = 0; kt < ktiles; kt++) {
    const int kb = kt * 32;
    // S = Q @ K^T : K fragments straight from global (B layout)
    v8f sc[2];
#pragma unroll
    for (int si = 0; si < 2; si++) {
      const _Float16* kr = kp + (size_t)(kb + si * 16 + ln) * HDIM;
      v16h kf0 = hcat(ld8h(kr + 16 * g),      ld8h(kr + 16 * g + 8));
      v16h kf1 = hcat(ld8h(kr + 32 + 16 * g), ld8h(kr + 40 + 16 * g));
      v8f z = zero8f();
      z = wmma_f16(qa[0], kf0, z);
      z = wmma_f16(qa[1], kf1, z);
      sc[si] = z;
    }
    // scale + causal mask + online softmax (row = r + 8*g, col = lane%16)
    float alpha[8];
#pragma unroll
    for (int r = 0; r < 8; r++) {
      const int qabs = q0 + r + 8 * g;
      float s0 = sc[0][r] * 0.125f;
      float s1 = sc[1][r] * 0.125f;
      if (kb + ln > qabs)      s0 = -1e30f;
      if (kb + 16 + ln > qabs) s1 = -1e30f;
      float mx = fmaxf(s0, s1);
#pragma unroll
      for (int off = 8; off > 0; off >>= 1) mx = fmaxf(mx, __shfl_xor(mx, off, 32));
      const float mn = fmaxf(mrow[r], mx);
      alpha[r] = __expf(mrow[r] - mn);
      mrow[r] = mn;
      const float p0 = __expf(s0 - mn);
      const float p1 = __expf(s1 - mn);
      sc[0][r] = p0; sc[1][r] = p1;
      float ls = p0 + p1;
#pragma unroll
      for (int off = 8; off > 0; off >>= 1) ls += __shfl_xor(ls, off, 32);
      lrow[r] = lrow[r] * alpha[r] + ls;
    }
    // rescale O, stage P (C layout -> LDS row-major, only layout round-trip)
#pragma unroll
    for (int nf = 0; nf < 4; nf++)
#pragma unroll
      for (int r = 0; r < 8; r++) o[nf][r] *= alpha[r];
#pragma unroll
    for (int si = 0; si < 2; si++)
#pragma unroll
      for (int r = 0; r < 8; r++)
        ps[(r + 8 * g) * 40 + si * 16 + ln] = (_Float16)sc[si][r];
    __asm__ volatile("s_wait_dscnt 0" ::: "memory");
    v16h pa;
    {
      const _Float16* pr = ps + ln * 40;
      pa = hcat(ld8h(pr + 16 * g), ld8h(pr + 16 * g + 8));
    }
    // O += P @ V : V fragments straight from global ([d][t] pre-transposed)
#pragma unroll
    for (int nf = 0; nf < 4; nf++) {
      const _Float16* vr = vp + (size_t)(nf * 16 + ln) * SEQ + kb + 16 * g;
      v16h vb = hcat(ld8h(vr), ld8h(vr + 8));
      o[nf] = wmma_f16(pa, vb, o[nf]);
    }
  }
  // y[b, q, h*64 + d] = O / l  (f16, feeds proj GEMM)
#pragma unroll
  for (int nf = 0; nf < 4; nf++) {
#pragma unroll
    for (int r = 0; r < 8; r++) {
      const int q = q0 + r + 8 * g;
      y[((size_t)b * SEQ + q) * DMODEL + h * HDIM + nf * 16 + ln] =
          (_Float16)(o[nf][r] / lrow[r]);
    }
  }
}

// ---------------------------------------------------------------------------
extern "C" void kernel_launch(void* const* d_in, const int* in_sizes, int n_in,
                              void* d_out, int out_size, void* d_ws, size_t ws_size,
                              hipStream_t stream) {
  const int*   idx    = (const int*)d_in[0];
  const float* wte    = (const float*)d_in[2];
  const float* wpe    = (const float*)d_in[3];
  const float* ln1_g  = (const float*)d_in[4];
  const float* ln1_b  = (const float*)d_in[5];
  const float* attn_w = (const float*)d_in[6];
  const float* attn_b = (const float*)d_in[7];
  const float* proj_w = (const float*)d_in[8];
  const float* proj_b = (const float*)d_in[9];
  const float* ln2_g  = (const float*)d_in[10];
  const float* ln2_b  = (const float*)d_in[11];
  const float* fc_w   = (const float*)d_in[12];
  const float* fc_b   = (const float*)d_in[13];
  const float* fc2_w  = (const float*)d_in[14];
  const float* fc2_b  = (const float*)d_in[15];
  const float* lnf_g  = (const float*)d_in[16];
  const float* lnf_b  = (const float*)d_in[17];
  const float* lm_w   = (const float*)d_in[18];
  float* out = (float*)d_out;

  const size_t BT = (size_t)BATCH * SEQ;  // 4096
  float*    x     = (float*)d_ws;                          // [BT][768] f32 residual
  _Float16* a16   = (_Float16*)(x + BT * DMODEL);          // LN outputs  f16
  _Float16* h2_16 = a16   + BT * DMODEL;                   // GELU out    f16
  _Float16* y16   = h2_16 + BT * 4 * DMODEL;               // attn out    f16
  _Float16* qh    = y16   + BT * DMODEL;                   // [b][h][t][64]
  _Float16* kh    = qh    + BT * DMODEL;                   // [b][h][t][64]
  _Float16* vth   = kh    + BT * DMODEL;                   // [b][h][64][t]
  _Float16* wqkvT = vth   + BT * DMODEL;                   // f16 transposed weights
  _Float16* wprojT= wqkvT + (size_t)NLAYER * 3 * DMODEL * DMODEL;
  _Float16* wfcT  = wprojT+ (size_t)NLAYER * DMODEL * DMODEL;
  _Float16* wfc2T = wfcT  + (size_t)NLAYER * 4 * DMODEL * DMODEL;
  _Float16* wlmT  = wfc2T + (size_t)NLAYER * 4 * DMODEL * DMODEL;  // [VPAD][768]

  const dim3 blk256(256), blk128(128);
  const int Mrows = (int)BT;

  // one-time (per launch) weight convert+transpose to f16 [N][K]
  for (int l = 0; l < NLAYER; l++) {
    convT_kernel<<<dim3(3 * DMODEL / 32, DMODEL / 32), blk256, 0, stream>>>(
        attn_w + (size_t)l * DMODEL * 3 * DMODEL, wqkvT + (size_t)l * 3 * DMODEL * DMODEL,
        DMODEL, 3 * DMODEL);
    convT_kernel<<<dim3(DMODEL / 32, DMODEL / 32), blk256, 0, stream>>>(
        proj_w + (size_t)l * DMODEL * DMODEL, wprojT + (size_t)l * DMODEL * DMODEL,
        DMODEL, DMODEL);
    convT_kernel<<<dim3(4 * DMODEL / 32, DMODEL / 32), blk256, 0, stream>>>(
        fc_w + (size_t)l * DMODEL * 4 * DMODEL, wfcT + (size_t)l * 4 * DMODEL * DMODEL,
        DMODEL, 4 * DMODEL);
    convT_kernel<<<dim3(DMODEL / 32, 4 * DMODEL / 32), blk256, 0, stream>>>(
        fc2_w + (size_t)l * 4 * DMODEL * DMODEL, wfc2T + (size_t)l * 4 * DMODEL * DMODEL,
        4 * DMODEL, DMODEL);
  }
  convT_kernel<<<dim3(VPAD / 32, DMODEL / 32), blk256, 0, stream>>>(
      lm_w, wlmT, DMODEL, VOCAB);

  embed_kernel<<<Mrows, blk256, 0, stream>>>(idx, wte, wpe, x);

  for (int l = 0; l < NLAYER; l++) {
    ln_kernel<<<Mrows, blk256, 0, stream>>>(x, ln1_g + l * DMODEL, ln1_b + l * DMODEL, a16);
    qkv_gemm_kernel<<<dim3(3 * DMODEL / 128, Mrows / 128), blk256, 0, stream>>>(
        a16, wqkvT + (size_t)l * 3 * DMODEL * DMODEL, attn_b + (size_t)l * 3 * DMODEL,
        qh, kh, vth);
    attn_kernel<<<dim3(SEQ / 64, BATCH * NHEAD), blk128, 0, stream>>>(qh, kh, vth, y16);
    gemm16_kernel<0, 1, 1, 0><<<dim3(DMODEL / 128, Mrows / 128), blk256, 0, stream>>>(
        y16, wprojT + (size_t)l * DMODEL * DMODEL, proj_b + (size_t)l * DMODEL,
        x, x, nullptr, DMODEL, DMODEL);
    ln_kernel<<<Mrows, blk256, 0, stream>>>(x, ln2_g + l * DMODEL, ln2_b + l * DMODEL, a16);
    gemm16_kernel<1, 0, 1, 1><<<dim3(4 * DMODEL / 128, Mrows / 128), blk256, 0, stream>>>(
        a16, wfcT + (size_t)l * 4 * DMODEL * DMODEL, fc_b + (size_t)l * 4 * DMODEL,
        nullptr, nullptr, h2_16, 4 * DMODEL, DMODEL);
    gemm16_kernel<0, 1, 1, 0><<<dim3(DMODEL / 128, Mrows / 128), blk256, 0, stream>>>(
        h2_16, wfc2T + (size_t)l * 4 * DMODEL * DMODEL, fc2_b + (size_t)l * DMODEL,
        x, x, nullptr, DMODEL, 4 * DMODEL);
  }

  ln_kernel<<<Mrows, blk256, 0, stream>>>(x, lnf_g, lnf_b, a16);
  gemm16_kernel<0, 0, 0, 0><<<dim3(VPAD / 128, Mrows / 128), blk256, 0, stream>>>(
      a16, wlmT, nullptr, nullptr, out, nullptr, VOCAB, DMODEL);
}
